// BaseModel_35356170781010
// MI455X (gfx1250) — matrix operand order repaired and checked by reading, not compiled, not run on hardware
//
#include <hip/hip_runtime.h>
#include <hip/hip_bf16.h>

// ---------------------------------------------------------------------------
// SPINN / TreeLSTM on gfx1250.
//   B=256, N=64, SIZE=512, D2=1024, T=127, G5=5*SIZE=2560, MAXD=65
// Per REDUCE step: lstm_in = [lh|rh](256x1024) @ [Wl;Wr](1024x2560) + bl
// GEMM: v_wmma_f32_16x16x32_bf16; A tile staged in LDS via
// global_load_async_to_lds_b128 (ASYNCcnt); B double-buffered in registers.
// ---------------------------------------------------------------------------

#define B_   256
#define N_   64
#define SZ_  512
#define D2_  1024
#define T_   127
#define G5_  2560
#define MAXD_ 65

typedef __attribute__((ext_vector_type(16))) __bf16 v16bf;
typedef __attribute__((ext_vector_type(8)))  float  v8f;

static __device__ __forceinline__ int imax(int a, int b) { return a > b ? a : b; }
static __device__ __forceinline__ int imin(int a, int b) { return a < b ? a : b; }
static __device__ __forceinline__ float sigm(float x) { return 1.0f / (1.0f + __expf(-x)); }

// ---------------------------------------------------------------------------
// prep: zero stack + ptr/bptr state (grid-stride)
// ---------------------------------------------------------------------------
__global__ void spinn_init(float* __restrict__ stack, int* __restrict__ ptrs /*4*B ints*/) {
    size_t total = (size_t)B_ * MAXD_ * D2_;
    for (size_t i = blockIdx.x * (size_t)blockDim.x + threadIdx.x; i < total;
         i += (size_t)gridDim.x * blockDim.x) {
        stack[i] = 0.0f;
    }
    int g = blockIdx.x * blockDim.x + threadIdx.x;
    if (g < 4 * B_) ptrs[g] = 0;
}

// ---------------------------------------------------------------------------
// pack [Wl;Wr] (f32, row-major 1024x2560) -> bf16 fragment-linear wpack.
// wpack[((nt*32 + kt)*32 + lane)*16 + i] = W[kt*32 + K(i)][nt*16 + (lane&15)]
// K(i<8) = (lane>>4)*8 + i, K(i>=8) = 16 + (lane>>4)*8 + (i-8) (same k order as A)
// ---------------------------------------------------------------------------
__global__ void spinn_packW(const float* __restrict__ Wl, const float* __restrict__ Wr,
                            __bf16* __restrict__ wpack) {
    int gid = blockIdx.x * blockDim.x + threadIdx.x;   // 160*32*32 = 163840
    if (gid >= (G5_ / 16) * 32 * 32) return;
    int lane = gid & 31;
    int kt   = (gid >> 5) & 31;
    int nt   = gid >> 10;
    int col  = nt * 16 + (lane & 15);
    int krow0 = kt * 32 + (lane >> 4) * 8;
    __bf16* dst = wpack + (size_t)gid * 16;
#pragma unroll
    for (int i = 0; i < 16; ++i) {
        int row = krow0 + ((i < 8) ? i : (8 + i));
        float v = (row < SZ_) ? Wl[(size_t)row * G5_ + col]
                              : Wr[(size_t)(row - SZ_) * G5_ + col];
        dst[i] = (__bf16)v;
    }
}

// ---------------------------------------------------------------------------
// gather: build bf16 A = [lh|rh] in fragment-linear layout from stack.
// apack[((mt*32 + kt)*32 + lane)*16 + i]; row = mt*16 + (lane&15);
//   K(i<8)  = kt*32 + (lane>>4)*8 + i
//   K(i>=8) = kt*32 + 16 + (lane>>4)*8 + (i-8)
// (ISA A-matrix 16x32 bf16 layout)
// ---------------------------------------------------------------------------
__global__ void spinn_gather(const float* __restrict__ stack, const int* __restrict__ ptr_in,
                             __bf16* __restrict__ apack) {
    int gid = blockIdx.x * blockDim.x + threadIdx.x;   // 16*32*32 = 16384
    if (gid >= 16 * 32 * 32) return;
    int lane = gid & 31;
    int kt   = (gid >> 5) & 31;
    int mt   = gid >> 10;
    int b     = mt * 16 + (lane & 15);
    int khalf = lane >> 4;
    int p = ptr_in[b];
    const float* left  = stack + ((size_t)b * MAXD_ + imax(p - 2, 0)) * D2_;
    const float* right = stack + ((size_t)b * MAXD_ + imax(p - 1, 0)) * D2_;
    __bf16* dst = apack + (size_t)gid * 16;
#pragma unroll
    for (int i = 0; i < 16; ++i) {
        int K = kt * 32 + ((i < 8) ? (khalf * 8 + i) : (16 + khalf * 8 + (i - 8)));
        float v = (K < SZ_) ? left[K] : right[K - SZ_];
        dst[i] = (__bf16)v;
    }
}

// ---------------------------------------------------------------------------
// gemm: lstm_in = A @ W + bl, WMMA bf16 -> f32.
// grid = (5, 16): blockIdx.y = M-tile (16 rows), blockIdx.x * 512 = col base.
// 8 waves/block, each wave: 16(M) x 64(N) = 4 accumulators, K loop of 32.
// A tile (16x1024 bf16 = 32KB) async-copied to dynamic LDS, shared by all
// 8 waves (ds_load_b128 in the loop). B fragments double-buffered from L2.
// Early-exit when no row in the M-tile does a REDUCE at step t.
// ---------------------------------------------------------------------------
__global__ __launch_bounds__(256) void spinn_gemm(
    const __bf16* __restrict__ apack, const __bf16* __restrict__ wpack,
    const float* __restrict__ bl, const int* __restrict__ trans,
    float* __restrict__ lstm, int t) {
    extern __shared__ __bf16 atile[];   // 16384 elements = 32 KB
    int mt = blockIdx.y;
    bool any_reduce = false;
    for (int r = 0; r < 16; ++r)
        any_reduce |= (trans[(mt * 16 + r) * T_ + t] == 1);
    if (!any_reduce) return;   // uniform across block (before any barrier)

    // --- async copy of the A tile into LDS (ASYNCcnt-tracked) ---
    {
        const __bf16* src = apack + (size_t)mt * 16384 + (size_t)threadIdx.x * 64;
        // dynamic-LDS generic address: low 32 bits == LDS byte offset
        uint32_t dOff = (uint32_t)(uintptr_t)atile + (uint32_t)threadIdx.x * 128u;
#pragma unroll
        for (int c = 0; c < 8; ++c) {
            asm volatile("global_load_async_to_lds_b128 %0, %1, off"
                         :: "v"(dOff + (uint32_t)(c * 16)), "v"(src + c * 8)
                         : "memory");
        }
        asm volatile("s_wait_asynccnt 0x0" ::: "memory");
    }
    __syncthreads();

    int wave = threadIdx.x >> 5;
    int lane = threadIdx.x & 31;
    int colBase = blockIdx.x * 512 + wave * 64;
    int nt0 = colBase >> 4;

    const __bf16* bP0 = wpack + ((size_t)(nt0 + 0) * 32 * 512) + (size_t)lane * 16;
    const __bf16* bP1 = wpack + ((size_t)(nt0 + 1) * 32 * 512) + (size_t)lane * 16;
    const __bf16* bP2 = wpack + ((size_t)(nt0 + 2) * 32 * 512) + (size_t)lane * 16;
    const __bf16* bP3 = wpack + ((size_t)(nt0 + 3) * 32 * 512) + (size_t)lane * 16;

    v8f acc0 = {}, acc1 = {}, acc2 = {}, acc3 = {};
    // register double-buffer for B fragments (wpack has a 4KB tail pad, so
    // the kt==31 lookahead overread is in-bounds of the workspace)
    v16bf b0 = *(const v16bf*)bP0;
    v16bf b1 = *(const v16bf*)bP1;
    v16bf b2 = *(const v16bf*)bP2;
    v16bf b3 = *(const v16bf*)bP3;
#pragma unroll
    for (int kt = 0; kt < 32; ++kt) {
        v16bf n0 = *(const v16bf*)(bP0 + (size_t)(kt + 1) * 512);
        v16bf n1 = *(const v16bf*)(bP1 + (size_t)(kt + 1) * 512);
        v16bf n2 = *(const v16bf*)(bP2 + (size_t)(kt + 1) * 512);
        v16bf n3 = *(const v16bf*)(bP3 + (size_t)(kt + 1) * 512);
        v16bf a  = *(const v16bf*)(atile + kt * 512 + lane * 16);   // ds_load_b128 x2
        acc0 = __builtin_amdgcn_wmma_f32_16x16x32_bf16(false, a, false, b0, (short)0, acc0, false, false);
        acc1 = __builtin_amdgcn_wmma_f32_16x16x32_bf16(false, a, false, b1, (short)0, acc1, false, false);
        acc2 = __builtin_amdgcn_wmma_f32_16x16x32_bf16(false, a, false, b2, (short)0, acc2, false, false);
        acc3 = __builtin_amdgcn_wmma_f32_16x16x32_bf16(false, a, false, b3, (short)0, acc3, false, false);
        b0 = n0; b1 = n1; b2 = n2; b3 = n3;
    }

    // C/D layout: VGPR r, lanes0-15 -> M=r, lanes16-31 -> M=r+8; col = lane&15
    int rowBase = mt * 16 + (lane >> 4) * 8;
    int c0 = colBase + (lane & 15);
#pragma unroll
    for (int r = 0; r < 8; ++r) {
        size_t row = (size_t)(rowBase + r) * G5_;
        lstm[row + c0 +  0] = acc0[r] + bl[c0 +  0];
        lstm[row + c0 + 16] = acc1[r] + bl[c0 + 16];
        lstm[row + c0 + 32] = acc2[r] + bl[c0 + 32];
        lstm[row + c0 + 48] = acc3[r] + bl[c0 + 48];
    }
}

// ---------------------------------------------------------------------------
// update: gates + stack write + ptr/bptr ping-pong update.
// One thread owns the element pair (j, j+512) of its batch row -> no races.
// ---------------------------------------------------------------------------
__global__ void spinn_update(const float* __restrict__ buffers, const int* __restrict__ trans,
                             const float* __restrict__ lstm, float* __restrict__ stack,
                             const int* __restrict__ ptr_in, const int* __restrict__ bptr_in,
                             int* __restrict__ ptr_out, int* __restrict__ bptr_out, int t) {
    int gid = blockIdx.x * blockDim.x + threadIdx.x;  // B*512
    if (gid >= B_ * SZ_) return;
    int b = gid >> 9;
    int j = gid & (SZ_ - 1);
    int tr  = trans[b * T_ + t];
    int p   = ptr_in[b];
    int bp  = bptr_in[b];
    if (j == 0) {
        ptr_out[b]  = p + ((tr == 0) ? 1 : ((tr == 1) ? -1 : 0));
        bptr_out[b] = bp + ((tr == 0) ? 1 : 0);
    }
    if (tr == 2) return;  // SKIP: no write
    float ih, ic;
    int wp;
    if (tr == 0) {  // SHIFT
        const float* src = buffers + ((size_t)b * N_ + imin(bp, N_ - 1)) * D2_;
        ih = src[j];
        ic = src[SZ_ + j];
        wp = p;
    } else {        // REDUCE
        const float* L = stack + ((size_t)b * MAXD_ + imax(p - 2, 0)) * D2_;
        const float* R = stack + ((size_t)b * MAXD_ + imax(p - 1, 0)) * D2_;
        float lc = L[SZ_ + j];
        float rc = R[SZ_ + j];
        const float* g = lstm + (size_t)b * G5_;
        float a  = g[j];
        float ii = g[SZ_ + j];
        float f1 = g[2 * SZ_ + j];
        float f2 = g[3 * SZ_ + j];
        float o  = g[4 * SZ_ + j];
        float c  = tanhf(a) * sigm(ii) + sigm(f1) * lc + sigm(f2) * rc;
        float h  = sigm(o) * tanhf(c);
        ih = h;
        ic = c;
        wp = imax(p - 2, 0);
    }
    float* dst = stack + ((size_t)b * MAXD_ + wp) * D2_;
    dst[j] = ih;
    dst[SZ_ + j] = ic;
}

// ---------------------------------------------------------------------------
// extract: out[b][j] = stack[b][max(ptr-1,0)][j], j < 512
// ---------------------------------------------------------------------------
__global__ void spinn_extract(const float* __restrict__ stack, const int* __restrict__ ptr_in,
                              float* __restrict__ out) {
    int gid = blockIdx.x * blockDim.x + threadIdx.x;
    if (gid >= B_ * SZ_) return;
    int b = gid >> 9;
    int j = gid & (SZ_ - 1);
    int p = imax(ptr_in[b] - 1, 0);
    out[gid] = stack[((size_t)b * MAXD_ + p) * D2_ + j];
}

// ---------------------------------------------------------------------------
extern "C" void kernel_launch(void* const* d_in, const int* in_sizes, int n_in,
                              void* d_out, int out_size, void* d_ws, size_t ws_size,
                              hipStream_t stream) {
    (void)in_sizes; (void)n_in; (void)out_size; (void)ws_size;
    const float* buffers = (const float*)d_in[0];
    const int*   trans   = (const int*)d_in[1];
    const float* Wl      = (const float*)d_in[2];
    const float* Wr      = (const float*)d_in[3];
    const float* bl      = (const float*)d_in[4];
    float* out = (float*)d_out;

    char* ws = (char*)d_ws;
    const size_t STACK_BYTES = (size_t)B_ * MAXD_ * D2_ * sizeof(float);   // 68,157,440
    const size_t WPACK_BYTES = (size_t)D2_ * G5_ * sizeof(__bf16);         //  5,242,880
    const size_t WPAD_BYTES  = 4096;   // tail pad for kt==31 lookahead overread
    const size_t APACK_BYTES = (size_t)B_ * D2_ * sizeof(__bf16);          //    524,288
    const size_t LSTM_BYTES  = (size_t)B_ * G5_ * sizeof(float);           //  2,621,440

    float*  stack = (float*)ws;                          ws += STACK_BYTES;
    __bf16* wpack = (__bf16*)ws;                         ws += WPACK_BYTES + WPAD_BYTES;
    __bf16* apack = (__bf16*)ws;                         ws += APACK_BYTES;
    float*  lstm  = (float*)ws;                          ws += LSTM_BYTES;
    int*    ptrs  = (int*)ws;  // 4*B ints: ptr[0],ptr[1],bptr[0],bptr[1]
    int* ptrBuf[2]  = { ptrs,          ptrs + B_ };
    int* bptrBuf[2] = { ptrs + 2 * B_, ptrs + 3 * B_ };

    spinn_init <<<2048, 256, 0, stream>>>(stack, ptrs);
    spinn_packW<<<(G5_ / 16) * 32 * 32 / 256, 256, 0, stream>>>(Wl, Wr, wpack);

    for (int t = 0; t < T_; ++t) {
        int* pIn  = ptrBuf[t & 1];
        int* pOut = ptrBuf[(t + 1) & 1];
        int* bIn  = bptrBuf[t & 1];
        int* bOut = bptrBuf[(t + 1) & 1];
        spinn_gather<<<16 * 32 * 32 / 256, 256, 0, stream>>>(stack, pIn, apack);
        spinn_gemm  <<<dim3(5, 16), 256, 32768, stream>>>(apack, wpack, bl, trans, lstm, t);
        spinn_update<<<B_ * SZ_ / 256, 256, 0, stream>>>(buffers, trans, lstm, stack,
                                                         pIn, bIn, pOut, bOut, t);
    }
    spinn_extract<<<B_ * SZ_ / 256, 256, 0, stream>>>(stack, ptrBuf[T_ & 1], out);
}
